// Attention_26044681683116
// MI455X (gfx1250) — compile-verified, matrix operand order checked by reading, not run on hardware
//
#include <hip/hip_runtime.h>
#include <hip/hip_bf16.h>

// Problem constants (fixed by the reference):
//   B=2, N=2048, D=1024, DHAT=512, H=16, F=32, G=64
#define BB   2
#define NN   2048
#define DD   1024
#define DHAT 512
#define HH   16
#define FF   32
#define GG   64
#define MM   (BB * NN)          // 4096 rows in all GEMMs
#define MBLK (MM / 16)          // 256 16-row blocks (all GEMMs share M=4096)

typedef __attribute__((ext_vector_type(16))) __bf16 v16bf;
typedef __attribute__((ext_vector_type(8)))  float  v8f;
typedef int v4i_gcc __attribute__((vector_size(16)));   // matches builtin "V4i"

// Async global->LDS staging (CDNA5 ASYNCcnt path), with safe fallback.
#if __has_builtin(__builtin_amdgcn_global_load_async_to_lds_b128) && \
    __has_builtin(__builtin_amdgcn_s_wait_asynccnt)
#define USE_ASYNC_LDS 1
#else
#define USE_ASYNC_LDS 0
#endif

__device__ __forceinline__ void copy16_g2l(__bf16* ldst, const __bf16* gsrc) {
#if USE_ASYNC_LDS
  // (global v4i* src, local v4i* dst, imm offset, imm cpol)
  __builtin_amdgcn_global_load_async_to_lds_b128(
      (__attribute__((address_space(1))) v4i_gcc*)gsrc,
      (__attribute__((address_space(3))) v4i_gcc*)ldst,
      /*offset=*/0, /*cpol=*/0);
#else
  *(uint4*)ldst = *(const uint4*)gsrc;
#endif
}

__device__ __forceinline__ void wait_async_le(int) {}
#if USE_ASYNC_LDS
#define WAIT_ASYNC(n) __builtin_amdgcn_s_wait_asynccnt(n)
#else
#define WAIT_ASYNC(n) ((void)0)
#endif

// WMMA 16-bit fragment position for element (kk in 0..31, rn in 0..15):
//   lanes 0-15 hold K in {0..7} (i=0..7) and {16..23} (i=8..15)
//   lanes16-31 hold K in {8..15}        and {24..31}
__device__ __forceinline__ void frag_pos(int kk, int rn, int& lane, int& i) {
  const int q = kk >> 3;                    // 0..3
  lane = rn + ((q & 1) ? 16 : 0);
  i    = (kk & 7) + ((q & 2) ? 8 : 0);
}

// ---------------------------------------------------------------------------
// Swizzle fp32 [R x C] row-major into bf16 WMMA-A fragment order:
//   dst[((kb*MBLK + mb)*32 + lane)*16 + i],  mb = row/16, kb = col/32
// ---------------------------------------------------------------------------
__global__ void swizzleA_f32_kernel(const float* __restrict__ src,
                                    __bf16* __restrict__ dst, int Kdim, int n) {
  int e = blockIdx.x * blockDim.x + threadIdx.x;
  int stride = gridDim.x * blockDim.x;
  for (; e < n; e += stride) {
    const int m = e / Kdim, k = e - m * Kdim;
    const int mb = m >> 4, r = m & 15;
    const int kb = k >> 5, kk = k & 31;
    int lane, i;
    frag_pos(kk, r, lane, i);
    dst[((size_t)(kb * MBLK + mb) * 32 + lane) * 16 + i] = (__bf16)src[e];
  }
}

// ---------------------------------------------------------------------------
// Swizzle fp32 weight [Kdim x Nout] row-major into bf16 WMMA-B fragment order:
//   dst[((kb*JB + j)*32 + lane)*16 + i],  j = n/16, kb = k/32, JB = Nout/16
// ---------------------------------------------------------------------------
__global__ void swizzleB_f32_kernel(const float* __restrict__ src,
                                    __bf16* __restrict__ dst, int Nout, int n) {
  int e = blockIdx.x * blockDim.x + threadIdx.x;
  int stride = gridDim.x * blockDim.x;
  const int JB = Nout >> 4;
  for (; e < n; e += stride) {
    const int k = e / Nout, nc = e - k * Nout;
    const int j = nc >> 4, nl = nc & 15;
    const int kb = k >> 5, kk = k & 31;
    int lane, i;
    frag_pos(kk, nl, lane, i);
    dst[((size_t)(kb * JB + j) * 32 + lane) * 16 + i] = (__bf16)src[e];
  }
}

// ---------------------------------------------------------------------------
// bf16 GEMM on pre-swizzled operands, fp32 accum via v_wmma_f32_16x16x32_bf16.
// Block: 256 threads (8 waves). Tile 128(M) x 64(N), K stepped by 32.
// 4-deep LDS ring, async staging issued 2 K-steps ahead. The pipeline tail is
// PEELED so the hot loop has a constant s_wait_asynccnt(6) and no
// data-dependent control flow; last two K-steps use wait(3)/wait(0).
// ONE barrier per K-step (ring-slot overwrite is separated from its last
// reader by an intervening barrier; ds reads retire before barrier signal).
// grid.x = Nout/64, grid.y = M/128.
// ---------------------------------------------------------------------------
template <bool PHI>
__global__ __launch_bounds__(256) void gemm_sw_kernel(
    const __bf16* __restrict__ Asw, const __bf16* __restrict__ Bsw,
    const float* __restrict__ bias, float* __restrict__ C,
    int Kdim, int Nout) {
  __shared__ __align__(32) __bf16 ldsA[4][8 * 512];   // 4 x 8KB ring
  __shared__ __align__(32) __bf16 ldsB[4][4 * 512];   // 4 x 4KB ring

  const int tid  = threadIdx.x;
  const int wave = tid >> 5;
  const int lane = tid & 31;
  const int mb0  = blockIdx.y * 8;      // first 16-row block of this tile
  const int j0   = blockIdx.x * 4;      // first 16-col block of this tile
  const int JB   = Nout >> 4;
  const int KB   = Kdim >> 5;

  auto stage = [&](int buf, int kb) {
    const __bf16* asrc = Asw + (size_t)(kb * MBLK + mb0) * 512;
    const __bf16* bsrc = Bsw + (size_t)(kb * JB + j0) * 512;
    copy16_g2l(&ldsA[buf][tid * 8],         asrc + tid * 8);
    copy16_g2l(&ldsA[buf][(tid + 256) * 8], asrc + (size_t)(tid + 256) * 8);
    copy16_g2l(&ldsB[buf][tid * 8],         bsrc + tid * 8);
  };

  v8f acc[4] = {};

  auto compute = [&](int kb) {
    __syncthreads();
    const int buf = kb & 3;
    const v16bf afrag = *(const v16bf*)&ldsA[buf][(wave * 32 + lane) * 16];
    v16bf bfrag[4];
    #pragma unroll
    for (int j = 0; j < 4; ++j)
      bfrag[j] = *(const v16bf*)&ldsB[buf][(j * 32 + lane) * 16];
    #pragma unroll
    for (int j = 0; j < 4; ++j)
      acc[j] = __builtin_amdgcn_wmma_f32_16x16x32_bf16(
          /*neg_a=*/false, afrag, /*neg_b=*/false, bfrag[j],
          /*c_mod=*/(short)0, acc[j], /*reuse_a=*/false, /*reuse_b=*/false);
  };

  // pipeline prologue: groups 0 and 1 in flight
  stage(0, 0);
  if (KB > 1) stage(1, 1);

  int kb = 0;
  // main loop: stage-ahead exists -> constant wait immediate, branch-free body
  for (; kb + 2 < KB; ++kb) {
    stage((kb + 2) & 3, kb + 2);
    WAIT_ASYNC(6);        // groups kb+1, kb+2 may remain in flight
    compute(kb);
  }
  // peeled tail: second-to-last K-step (group kb+1 still in flight)
  if (kb + 1 < KB) {
    WAIT_ASYNC(3);
    compute(kb);
    ++kb;
  }
  // peeled tail: last K-step
  WAIT_ASYNC(0);
  compute(kb);

  // Epilogue: bias (+ phi = elu+1), store fp32 in natural [M x Nout] layout.
  // C/D layout: VGPR r -> (M = r + (lane>=16 ? 8 : 0), N = lane & 15)
  const int mbase = (mb0 + wave) * 16 + ((lane >= 16) ? 8 : 0);
  #pragma unroll
  for (int j = 0; j < 4; ++j) {
    const int n = (j0 + j) * 16 + (lane & 15);
    const float bn = bias[n];
    #pragma unroll
    for (int r = 0; r < 8; ++r) {
      float v = acc[j][r] + bn;
      if (PHI) v = (v > 0.0f) ? (v + 1.0f) : __expf(v);
      C[(size_t)(mbase + r) * Nout + n] = v;
    }
  }
}

// ---------------------------------------------------------------------------
// Causal linear-attention scan. One wave per (b,h); lane g keeps state rows
// S[g][0..31] and S[g+32][0..31] plus the duplicated K-cumsum in registers.
// Q[t,f], K[t,f] broadcast with wave32 shuffles; no LDS, no barriers.
// Output written as bf16 DIRECTLY in swizzled-A fragment order so the final
// WMMA GEMM consumes it with zero extra passes.
// ---------------------------------------------------------------------------
__global__ __launch_bounds__(32) void linattn_scan_kernel(
    const float* __restrict__ Q, const float* __restrict__ K,
    const float* __restrict__ V, __bf16* __restrict__ Osw) {
  const int bh = blockIdx.x;
  const int b  = bh / HH;
  const int h  = bh % HH;
  const int lane = threadIdx.x;   // g0 = lane, g1 = lane + 32

  float S0[FF], S1[FF], zs[FF];
  #pragma unroll
  for (int f = 0; f < FF; ++f) { S0[f] = 0.0f; S1[f] = 0.0f; zs[f] = 0.0f; }

  const float* Qb = Q + (size_t)b * NN * DHAT + h * FF;
  const float* Kb = K + (size_t)b * NN * DHAT + h * FF;
  const float* Vb = V + (size_t)b * NN * DD   + h * GG;

  // Swizzled store geometry (attn is the A matrix of the final GEMM):
  //   k-column = h*64 + g  ->  kb = h*2 + (g>=32), kk = g & 31 = lane
  int la, ii;
  frag_pos(lane, 0, la, ii);                 // r added per-timestep below
  const size_t kb0base = (size_t)(h * 2)     * MBLK * 512;
  const size_t kb1base = (size_t)(h * 2 + 1) * MBLK * 512;

  for (int t = 0; t < NN; ++t) {
    const float q  = Qb[(size_t)t * DHAT + lane];
    const float k  = Kb[(size_t)t * DHAT + lane];
    const float v0 = Vb[(size_t)t * DD + lane];
    const float v1 = Vb[(size_t)t * DD + 32 + lane];

    float acc0 = 0.0f, acc1 = 0.0f, denom = 0.0f;
    #pragma unroll
    for (int f = 0; f < FF; ++f) {
      const float kf = __shfl(k, f, 32);
      const float qf = __shfl(q, f, 32);
      zs[f] += kf;                 // cumsum(K) includes current t
      denom += qf * zs[f];
      S0[f] += kf * v0;            // kv state update (includes current t)
      S1[f] += kf * v1;
      acc0  += qf * S0[f];
      acc1  += qf * S1[f];
    }
    const float Z = 1.0f / (denom + 1e-6f);

    const int m  = b * NN + t;
    const int mb = m >> 4;
    const int r  = m & 15;
    const size_t frag = ((size_t)mb * 32 + (la + r)) * 16 + ii;
    Osw[kb0base + frag] = (__bf16)(acc0 * Z);
    Osw[kb1base + frag] = (__bf16)(acc1 * Z);
  }
}

// ---------------------------------------------------------------------------
// Launch
// ---------------------------------------------------------------------------
extern "C" void kernel_launch(void* const* d_in, const int* in_sizes, int n_in,
                              void* d_out, int out_size, void* d_ws, size_t ws_size,
                              hipStream_t stream) {
  (void)in_sizes; (void)n_in; (void)out_size; (void)ws_size;

  const float* x  = (const float*)d_in[0];
  const float* wq = (const float*)d_in[1];
  const float* bq = (const float*)d_in[2];
  const float* wk = (const float*)d_in[3];
  const float* bk = (const float*)d_in[4];
  const float* wv = (const float*)d_in[5];
  const float* bv = (const float*)d_in[6];
  const float* wo = (const float*)d_in[7];
  const float* bo = (const float*)d_in[8];
  float* out = (float*)d_out;

  // workspace carve-up (256B aligned)
  char* ws = (char*)d_ws;
  size_t off = 0;
  auto carve = [&](size_t bytes) -> char* {
    char* p = ws + off;
    off = (off + bytes + 255) & ~(size_t)255;
    return p;
  };
  __bf16* xsw    = (__bf16*)carve((size_t)MM * DD * 2);
  __bf16* wqsw   = (__bf16*)carve((size_t)DD * DHAT * 2);
  __bf16* wksw   = (__bf16*)carve((size_t)DD * DHAT * 2);
  __bf16* wvsw   = (__bf16*)carve((size_t)DD * DD * 2);
  __bf16* wosw   = (__bf16*)carve((size_t)DD * DD * 2);
  float*  Qf     = (float*)carve((size_t)MM * DHAT * 4);
  float*  Kf     = (float*)carve((size_t)MM * DHAT * 4);
  float*  Vf     = (float*)carve((size_t)MM * DD * 4);
  __bf16* attnsw = (__bf16*)carve((size_t)MM * DD * 2);

  const int blk = 256;

  // 1) convert + swizzle into WMMA fragment order
  {
    int n;
    n = MM * DD;
    swizzleA_f32_kernel<<<(n + blk - 1) / blk, blk, 0, stream>>>(x, xsw, DD, n);
    n = DD * DHAT;
    swizzleB_f32_kernel<<<(n + blk - 1) / blk, blk, 0, stream>>>(wq, wqsw, DHAT, n);
    swizzleB_f32_kernel<<<(n + blk - 1) / blk, blk, 0, stream>>>(wk, wksw, DHAT, n);
    n = DD * DD;
    swizzleB_f32_kernel<<<(n + blk - 1) / blk, blk, 0, stream>>>(wv, wvsw, DD, n);
    swizzleB_f32_kernel<<<(n + blk - 1) / blk, blk, 0, stream>>>(wo, wosw, DD, n);
  }

  // 2) projections: Q = phi(x@wq+bq), K = phi(x@wk+bk), V = x@wv+bv
  {
    dim3 tb(256);
    dim3 gQ(DHAT / 64, MM / 128);
    gemm_sw_kernel<true ><<<gQ, tb, 0, stream>>>(xsw, wqsw, bq, Qf, DD, DHAT);
    gemm_sw_kernel<true ><<<gQ, tb, 0, stream>>>(xsw, wksw, bk, Kf, DD, DHAT);
    dim3 gV(DD / 64, MM / 128);
    gemm_sw_kernel<false><<<gV, tb, 0, stream>>>(xsw, wvsw, bv, Vf, DD, DD);
  }

  // 3) causal linear-attention scan (one wave per (b,h)), writes swizzled attn
  linattn_scan_kernel<<<BB * HH, 32, 0, stream>>>(Qf, Kf, Vf, attnsw);

  // 4) output projection: out = attn @ wo + bo
  {
    dim3 tb(256);
    dim3 gO(DD / 64, MM / 128);
    gemm_sw_kernel<false><<<gO, tb, 0, stream>>>(attnsw, wosw, bo, out, DD, DD);
  }
}
